// MultiHeadedAttention_7267084665474
// MI455X (gfx1250) — matrix operand order repaired and checked by reading, not compiled, not run on hardware
//
#include <hip/hip_runtime.h>

#define Bn   4
#define Sn   2048
#define Hn   12
#define DKn  64
#define Dn   768
#define Mn   (Bn * Sn)          // 8192
#define LDP  40                  // LDS row pitch (halves) for P transpose buffer

typedef __attribute__((ext_vector_type(16))) _Float16 v16h;
typedef __attribute__((ext_vector_type(8)))  _Float16 v8h;
typedef __attribute__((ext_vector_type(8)))  float    v8f;
typedef __attribute__((ext_vector_type(4)))  unsigned int u32x4;
typedef __attribute__((ext_vector_type(8)))  int          i32x8;
typedef __attribute__((ext_vector_type(4)))  int          i32x4;

// ---------------------------------------------------------------- WMMA helpers

__device__ __forceinline__ v8f wmma_f16(v16h a, v16h b, v8f c) {
  // D = A(16x32 f16) * B(32x16 f16) + C(16x16 f32)
  return __builtin_amdgcn_wmma_f32_16x16x32_f16(
      /*neg_a=*/false, a, /*neg_b=*/false, b,
      /*c_mod=*/(short)0, c, /*reuse_a=*/false, /*reuse_b=*/false);
}

// A-fragment (16x32 f16) from row-major [*, lda] f16 matrix.
__device__ __forceinline__ v16h load_a_frag(const _Float16* __restrict__ A,
                                            int lda, int m0, int k0, int lane) {
  const int row = m0 + (lane & 15);
  const _Float16* p = A + (size_t)row * lda + k0 + ((lane & 16) ? 16 : 0);
  v8h lo = *(const v8h*)(p);
  v8h hi = *(const v8h*)(p + 8);
  v16h r;
#pragma unroll
  for (int e = 0; e < 8; ++e) { r[e] = lo[e]; r[e + 8] = hi[e]; }
  return r;
}

// B-fragment (32x16 f16) where B(k,n) = Bt[n][k] (Bt row-major [*, ldb]).
__device__ __forceinline__ v16h load_b_frag(const _Float16* __restrict__ Bt,
                                            int ldb, int n0, int k0, int lane) {
  const int row = n0 + (lane & 15);
  return *(const v16h*)(Bt + (size_t)row * ldb + k0 + ((lane & 16) ? 16 : 0));
}

// Row reductions across the 16-lane half-groups of the C-layout.
__device__ __forceinline__ float rmax16(float x) {
#pragma unroll
  for (int m = 1; m < 16; m <<= 1) x = fmaxf(x, __shfl_xor(x, m, 32));
  return x;
}
__device__ __forceinline__ float rsum16(float x) {
#pragma unroll
  for (int m = 1; m < 16; m <<= 1) x += __shfl_xor(x, m, 32);
  return x;
}

// ---------------------------------------------------------------- TDM helper
// 2D tile load Global->LDS via the Tensor Data Mover (D# per ISA ch.8).
// dims/strides in elements (data_size = 2 bytes), addresses in bytes.
__device__ __forceinline__ void tdm_load_2d(unsigned lds_addr,
                                            unsigned long long gaddr,
                                            unsigned tensor_dim0,
                                            unsigned tensor_dim1,
                                            unsigned tile_dim0,
                                            unsigned tile_dim1,
                                            unsigned long long stride0) {
  u32x4 g0;
  g0[0] = 1u;                                             // count=1, user mode
  g0[1] = lds_addr;                                       // lds_addr [63:32]
  g0[2] = (unsigned)(gaddr & 0xffffffffull);              // global_addr lo
  g0[3] = ((unsigned)((gaddr >> 32) & 0x01ffffffull))     // global_addr hi
          | (2u << 30);                                   // type = 2 (image)
  i32x8 g1;
  g1[0] = (int)(1u << 16);                                // data_size = 2B
  g1[1] = (int)((tensor_dim0 & 0xffffu) << 16);           // dim0 [79:48]
  g1[2] = (int)((tensor_dim0 >> 16) | ((tensor_dim1 & 0xffffu) << 16));
  g1[3] = (int)((tensor_dim1 >> 16) | (tile_dim0 << 16)); // tile_dim0 [127:112]
  g1[4] = (int)(tile_dim1);                               // tile_dim1, tile_dim2=0
  g1[5] = (int)(stride0 & 0xffffffffull);                 // dim0_stride [207:160]
  g1[6] = (int)((stride0 >> 32) & 0xffffull);             // dim1_stride = 0
  g1[7] = 0;
  i32x4 gz = {0, 0, 0, 0};
#if __clang_major__ >= 23
  i32x8 gz8 = {0, 0, 0, 0, 0, 0, 0, 0};
  __builtin_amdgcn_tensor_load_to_lds(g0, g1, gz, gz, gz8, 0);
#else
  __builtin_amdgcn_tensor_load_to_lds(g0, g1, gz, gz, 0);
#endif
}

// ---------------------------------------------------------------- pack kernels

__global__ void pack_f16_kernel(const float* __restrict__ src,
                                _Float16* __restrict__ dst, int n) {
  int i = blockIdx.x * blockDim.x + threadIdx.x;
  if (i < n) dst[i] = (_Float16)src[i];
}

__global__ void pack_wt_kernel(const float* __restrict__ W,
                               _Float16* __restrict__ Wt) {
  int i = blockIdx.x * blockDim.x + threadIdx.x;
  if (i < Dn * Dn) {
    int nn = i / Dn, kk = i - nn * Dn;
    Wt[i] = (_Float16)W[kk * Dn + nn];
  }
}

// ---------------------------------------------------------------- GEMM + epilogue
//   EPI 0 : fp32 store to out32 (final output projection)
//   EPI 1 : RoPE then f16 store to [B,H,S,DK]         (Q and K)
//   EPI 3 : f16 store transposed to [B,H,DK,S]        (V)
template <int EPI>
__global__ __launch_bounds__(256)
void gemm_wmma_kernel(const _Float16* __restrict__ A,
                      const _Float16* __restrict__ Bt,
                      const float* __restrict__ bias,
                      _Float16* __restrict__ o16,
                      float* __restrict__ o32) {
  const int lane = threadIdx.x & 31;
  const int w    = threadIdx.x >> 5;      // 8 waves: 2 (M) x 4 (N)
  const int wm   = w & 1;
  const int wn   = w >> 1;
  const int m0   = blockIdx.x * 64 + wm * 32;
  const int n0   = blockIdx.y * 128 + wn * 32;

  v8f zero = {};
  v8f acc[2][2];
  acc[0][0] = zero; acc[0][1] = zero; acc[1][0] = zero; acc[1][1] = zero;

  for (int k0 = 0; k0 < Dn; k0 += 32) {
    __builtin_prefetch(A + (size_t)(m0 + (lane & 15)) * Dn + k0 + 64, 0, 1);
    __builtin_prefetch(Bt + (size_t)(n0 + (lane & 15)) * Dn + k0 + 64, 0, 1);
    v16h a0 = load_a_frag(A, Dn, m0,      k0, lane);
    v16h a1 = load_a_frag(A, Dn, m0 + 16, k0, lane);
    v16h b0 = load_b_frag(Bt, Dn, n0,      k0, lane);
    v16h b1 = load_b_frag(Bt, Dn, n0 + 16, k0, lane);
    acc[0][0] = wmma_f16(a0, b0, acc[0][0]);
    acc[0][1] = wmma_f16(a0, b1, acc[0][1]);
    acc[1][0] = wmma_f16(a1, b0, acc[1][0]);
    acc[1][1] = wmma_f16(a1, b1, acc[1][1]);
  }

  const int ln  = lane & 15;
  const int mup = (lane & 16) ? 8 : 0;
#pragma unroll
  for (int i = 0; i < 2; ++i) {
#pragma unroll
    for (int j = 0; j < 2; ++j) {
#pragma unroll
      for (int r = 0; r < 8; ++r) {
        const int m = m0 + i * 16 + mup + r;   // global row (= b*S + s)
        const int n = n0 + j * 16 + ln;        // global col (= h*64 + dk)
        float v = acc[i][j][r] + bias[n];
        if constexpr (EPI == 0) {
          o32[(size_t)m * Dn + n] = v;
        } else if constexpr (EPI == 3) {
          const int b = m >> 11, s = m & (Sn - 1);
          const int h = n >> 6, dk = n & 63;
          o16[(((size_t)(b * Hn + h) * DKn + dk) * Sn) + s] = (_Float16)v;
        } else {  // EPI == 1: RoPE
          const float part = __shfl_xor(v, 1, 32);   // pair partner (n ^ 1)
          const int b = m >> 11, s = m & (Sn - 1);
          const int h = n >> 6, dk = n & 63;
          const float theta = __powf(10000.0f, -(float)(dk & ~1) * (1.0f / 64.0f));
          const float ang = (float)s * theta;
          float sn, cs;
          __sincosf(ang, &sn, &cs);
          const float x2 = (dk & 1) ? part : -part;
          const float rv = v * cs + x2 * sn;
          o16[((size_t)(b * Hn + h) * Sn + s) * DKn + dk] = (_Float16)rv;
        }
      }
    }
  }
}

// ---------------------------------------------------------------- flash attention
// One block = 8 waves = one 128-query stripe of one (b,h). K/V chunks are
// staged into double-buffered LDS by the Tensor Data Mover: the TDM for
// chunk i+1 is issued right after chunk i becomes visible, so DMA overlaps
// the 8 WMMAs/wave of compute. TENSORcnt is the producer-side split barrier.
__global__ __launch_bounds__(256)
void flash_attn_kernel(const _Float16* __restrict__ Qh,   // [B,H,S,DK]
                       const _Float16* __restrict__ Kh,   // [B,H,S,DK]
                       const _Float16* __restrict__ Vt,   // [B,H,DK,S]
                       const unsigned char* __restrict__ mask,  // [B,1,S,S]
                       _Float16* __restrict__ Oh) {       // [B*S, H*DK]
  __shared__ __align__(16) _Float16 ksm[2][32 * DKn];     // K chunk (32 keys x 64)
  __shared__ __align__(16) _Float16 vsm[2][DKn * 32];     // V chunk (64 dk x 32 keys)
  __shared__ __align__(16) _Float16 pb[8][16 * LDP];      // per-wave P transpose

  const int lane = threadIdx.x & 31;
  const int wid  = threadIdx.x >> 5;
  const int bh   = blockIdx.x >> 4;                // 16 blocks per (b,h)
  const int b    = bh / Hn;
  const int h    = bh % Hn;
  const int qb   = (blockIdx.x & 15) * 128 + wid * 16;

  const _Float16* Qp = Qh + (size_t)(b * Hn + h) * Sn * DKn;
  const _Float16* Kp = Kh + (size_t)(b * Hn + h) * Sn * DKn;
  const _Float16* Vp = Vt + (size_t)(b * Hn + h) * DKn * Sn;
  const unsigned char* mp = mask + (size_t)b * Sn * Sn;

  const int ln  = lane & 15;
  const int mup = (lane & 16) ? 8 : 0;

  const v16h aq0 = load_a_frag(Qp, DKn, qb, 0, lane);
  const v16h aq1 = load_a_frag(Qp, DKn, qb, 32, lane);

  v8f zero = {};
  v8f o[4];
#pragma unroll
  for (int n = 0; n < 4; ++n) o[n] = zero;
  float rm[8], rs[8], f[8];
#pragma unroll
  for (int r = 0; r < 8; ++r) { rm[r] = -3.0e38f; rs[r] = 0.0f; }

  // ---- prologue: stage chunk 0 into buffer 0
  if (wid == 0) {
    tdm_load_2d((unsigned)(size_t)&ksm[0][0],
                (unsigned long long)(size_t)Kp, DKn, Sn, DKn, 32, DKn);
    tdm_load_2d((unsigned)(size_t)&vsm[0][0],
                (unsigned long long)(size_t)Vp, Sn, DKn, 32, DKn, Sn);
  }

  int p = 0;
  for (int kt = 0; kt < Sn; kt += 32, p ^= 1) {
    // ---- publish chunk kt (producer waits TENSORcnt, then block barrier)
    if (wid == 0) __builtin_amdgcn_s_wait_tensorcnt(0);
    __syncthreads();

    // ---- prefetch chunk kt+32 into the other buffer while we compute
    if (wid == 0 && kt + 32 < Sn) {
      tdm_load_2d((unsigned)(size_t)&ksm[p ^ 1][0],
                  (unsigned long long)(size_t)(Kp + (size_t)(kt + 32) * DKn),
                  DKn, Sn, DKn, 32, DKn);
      tdm_load_2d((unsigned)(size_t)&vsm[p ^ 1][0],
                  (unsigned long long)(size_t)(Vp + kt + 32),
                  Sn, DKn, 32, DKn, Sn);
    }

    // ---- scores: two 16x16 tiles over this 32-key chunk (B-frags from LDS)
    v8f s0 = zero, s1 = zero;
    v16h bk;
    bk = load_b_frag(ksm[p], DKn, 0,  0,  lane); s0 = wmma_f16(aq0, bk, s0);
    bk = load_b_frag(ksm[p], DKn, 0,  32, lane); s0 = wmma_f16(aq1, bk, s0);
    bk = load_b_frag(ksm[p], DKn, 16, 0,  lane); s1 = wmma_f16(aq0, bk, s1);
    bk = load_b_frag(ksm[p], DKn, 16, 32, lane); s1 = wmma_f16(aq1, bk, s1);

    // ---- online softmax (per C-layout row)
#pragma unroll
    for (int r = 0; r < 8; ++r) {
      const int qrow = qb + mup + r;
      float v0 = s0[r] * 0.125f;   // 1/sqrt(64)
      float v1 = s1[r] * 0.125f;
      if (!mp[(size_t)qrow * Sn + kt + ln])      v0 = -1.0e9f;
      if (!mp[(size_t)qrow * Sn + kt + 16 + ln]) v1 = -1.0e9f;
      const float mnew = fmaxf(rm[r], rmax16(fmaxf(v0, v1)));
      f[r] = __expf(rm[r] - mnew);
      rm[r] = mnew;
      const float p0 = __expf(v0 - mnew);
      const float p1 = __expf(v1 - mnew);
      rs[r] = rs[r] * f[r] + rsum16(p0 + p1);
      pb[wid][(mup + r) * LDP + ln]      = (_Float16)p0;
      pb[wid][(mup + r) * LDP + 16 + ln] = (_Float16)p1;
    }
    // per-wave DS ordering: make the cross-lane P tile visible before reload
    asm volatile("s_wait_dscnt 0" ::: "memory");

    // ---- reload P as a 16x32 A-fragment (transpose via LDS)
    v16h ap;
    {
      const _Float16* pp = &pb[wid][(lane & 15) * LDP + ((lane & 16) ? 16 : 0)];
      v8h lo = *(const v8h*)(pp);
      v8h hi = *(const v8h*)(pp + 8);
#pragma unroll
      for (int e = 0; e < 8; ++e) { ap[e] = lo[e]; ap[e + 8] = hi[e]; }
    }

    // ---- O(16x64) += P(16x32) @ V(32x64), rescaling running accumulators
#pragma unroll
    for (int n = 0; n < 4; ++n) {
      const v16h bv = load_b_frag(vsm[p], 32, n * 16, 0, lane);
#pragma unroll
      for (int r = 0; r < 8; ++r) o[n][r] *= f[r];
      o[n] = wmma_f16(ap, bv, o[n]);
    }
    __syncthreads();   // all waves done with buffer p before it is refilled
  }

  // ---- normalize and store f16 output rows [b*S + q, h*64 + dk]
#pragma unroll
  for (int n = 0; n < 4; ++n) {
#pragma unroll
    for (int r = 0; r < 8; ++r) {
      const float val = o[n][r] / rs[r];
      const int qrow = qb + mup + r;
      Oh[((size_t)(b * Sn + qrow)) * Dn + h * DKn + n * 16 + ln] = (_Float16)val;
    }
  }
}

// ---------------------------------------------------------------- launch

extern "C" void kernel_launch(void* const* d_in, const int* in_sizes, int n_in,
                              void* d_out, int out_size, void* d_ws, size_t ws_size,
                              hipStream_t stream) {
  (void)in_sizes; (void)n_in; (void)out_size; (void)ws_size;
  const float* q    = (const float*)d_in[0];
  const float* k    = (const float*)d_in[1];
  const float* v    = (const float*)d_in[2];
  const unsigned char* mask = (const unsigned char*)d_in[3];
  const float* Wq = (const float*)d_in[4];  const float* bq = (const float*)d_in[5];
  const float* Wk = (const float*)d_in[6];  const float* bk = (const float*)d_in[7];
  const float* Wv = (const float*)d_in[8];  const float* bv = (const float*)d_in[9];
  const float* Wo = (const float*)d_in[10]; const float* bo = (const float*)d_in[11];

  char* ws = (char*)d_ws;
  size_t off = 0;
  auto take = [&](size_t bytes) {
    size_t o = off;
    off = (off + bytes + 255) & ~(size_t)255;
    return o;
  };
  const size_t MD = (size_t)Mn * Dn;      // 6,291,456
  const size_t DD = (size_t)Dn * Dn;

  _Float16* Xq  = (_Float16*)(ws + take(MD * 2));
  _Float16* Xk  = (_Float16*)(ws + take(MD * 2));
  _Float16* Xv  = (_Float16*)(ws + take(MD * 2));
  _Float16* Wqt = (_Float16*)(ws + take(DD * 2));
  _Float16* Wkt = (_Float16*)(ws + take(DD * 2));
  _Float16* Wvt = (_Float16*)(ws + take(DD * 2));
  _Float16* Wot = (_Float16*)(ws + take(DD * 2));
  _Float16* Qh  = (_Float16*)(ws + take(MD * 2));
  _Float16* Kh  = (_Float16*)(ws + take(MD * 2));
  _Float16* Vt  = (_Float16*)(ws + take(MD * 2));
  _Float16* Oh  = (_Float16*)(ws + take(MD * 2));

  const int packBlocks = (int)((MD + 255) / 256);
  pack_f16_kernel<<<packBlocks, 256, 0, stream>>>(q, Xq, (int)MD);
  pack_f16_kernel<<<packBlocks, 256, 0, stream>>>(k, Xk, (int)MD);
  pack_f16_kernel<<<packBlocks, 256, 0, stream>>>(v, Xv, (int)MD);
  const int wtBlocks = (int)((DD + 255) / 256);
  pack_wt_kernel<<<wtBlocks, 256, 0, stream>>>(Wq, Wqt);
  pack_wt_kernel<<<wtBlocks, 256, 0, stream>>>(Wk, Wkt);
  pack_wt_kernel<<<wtBlocks, 256, 0, stream>>>(Wv, Wvt);
  pack_wt_kernel<<<wtBlocks, 256, 0, stream>>>(Wo, Wot);

  dim3 ggrid(Mn / 64, Dn / 128);   // 128 x 6
  gemm_wmma_kernel<1><<<ggrid, 256, 0, stream>>>(Xq, Wqt, bq, Qh, nullptr);
  gemm_wmma_kernel<1><<<ggrid, 256, 0, stream>>>(Xk, Wkt, bk, Kh, nullptr);
  gemm_wmma_kernel<3><<<ggrid, 256, 0, stream>>>(Xv, Wvt, bv, Vt, nullptr);

  flash_attn_kernel<<<Bn * Hn * (Sn / 128), 256, 0, stream>>>(
      Qh, Kh, Vt, mask, Oh);

  gemm_wmma_kernel<0><<<ggrid, 256, 0, stream>>>(Oh, Wot, bo, nullptr,
                                                 (float*)d_out);
}
